// GPT_386547056811
// MI455X (gfx1250) — matrix operand order repaired and checked by reading, not compiled, not run on hardware
//
#include <hip/hip_runtime.h>
#include <hip/hip_bf16.h>
#include <math.h>

// ---------------- Problem constants (GPT-2 small-ish, B=4, T=1024) ----------
#define V_SZ     50257
#define T_SEQ    1024
#define D_MODEL  768
#define H_HEADS  12
#define L_LAYERS 12
#define HS_HEAD  64
#define B_BATCH  4
#define EPS_LN   1e-5f
#define BT       (B_BATCH * T_SEQ)   // 4096 rows

// Async global->LDS copy path (gfx1250 GLOBAL_LOAD_ASYNC_TO_LDS_B128, ASYNCcnt)
#if defined(__has_builtin)
#  if __has_builtin(__builtin_amdgcn_global_load_async_to_lds_b128) && \
      __has_builtin(__builtin_amdgcn_s_wait_asynccnt)
#    define USE_ASYNC_LDS 1
#  endif
#endif
#ifndef USE_ASYNC_LDS
#  define USE_ASYNC_LDS 0
#endif

// ---------------- WMMA types -------------------------------------------------
typedef __attribute__((ext_vector_type(16))) __bf16 bf16x16;
typedef __attribute__((ext_vector_type(8)))  __bf16 bf16x8;
typedef __attribute__((ext_vector_type(8)))  float  floatx8;
typedef int v4i_vs __attribute__((__vector_size__(16)));            // b128 unit
typedef __attribute__((address_space(3))) v4i_vs* lds_v4i_p;        // LDS b128 ptr

static __device__ __forceinline__ unsigned short f2bf(float f) {
  // fp32 -> bf16 round-to-nearest-even
  unsigned int u = __builtin_bit_cast(unsigned int, f);
  u += 0x7FFFu + ((u >> 16) & 1u);
  return (unsigned short)(u >> 16);
}

// ---------------- Generic tiled WMMA GEMM -----------------------------------
// C[M,N] = A[M,K] @ B[K,N] (+ bias[n]) (ReLU) (+ res[M,N])
// fp32 in/out, bf16 WMMA 16x16x32 compute, fp32 accumulate.
// Block tile: 64x128, BK=32. 8 waves; each wave = one 16-row strip x 64 cols
// (1 A fragment reused over 4 B fragments -> 4 WMMAs per k-step per wave).
#define BM 64
#define BN 128
#define BK 32
#define AST 56   // A LDS row stride in ushorts (112 B: 16B-aligned, bank-spread)
#define BST 56   // B LDS row stride in ushorts (transposed [n][k])

__global__ __launch_bounds__(256)
void gemm_bf16_wmma(const float* __restrict__ A, const float* __restrict__ Bm,
                    const float* __restrict__ bias, const float* __restrict__ res,
                    float* __restrict__ C, int M, int N, int K, int relu)
{
  __shared__ unsigned short As[BM][AST];   // [m][k]   7.0 KB
  __shared__ unsigned short Bs[BN][BST];   // [n][k]  14.0 KB (transposed)

  const int tid  = threadIdx.x;
  const int lane = tid & 31;
  const int wave = tid >> 5;               // 0..7
  const int rowBase = blockIdx.y * BM;
  const int colBase = blockIdx.x * BN;

  const int tr  = wave & 3;                // subtile row 0..3 (16 rows each)
  const int tcb = (wave >> 2) * 64;        // wave's 64-col strip within tile

  const bool fullTile = (rowBase + BM <= M) && (colBase + BN <= N);

  floatx8 acc[4] = {floatx8{}, floatx8{}, floatx8{}, floatx8{}};

  // per-lane fragment addresses (contiguous k order, same for A and B)
  const int ar  = tr * 16 + (lane & 15);
  const int akh = (lane >> 4) * 16;        // A: lanes0-15 K 0..15, 16-31: K 16..31
  const int bkh = (lane >> 4) * 16;        // B: lanes0-15 K 0..15, 16-31: K 16..31
  const bf16x8* aptr0 = (const bf16x8*)&As[ar][akh];
  const bf16x8* aptr1 = (const bf16x8*)&As[ar][akh + 8];
  const bf16x8* bptr[4];
  #pragma unroll
  for (int tt = 0; tt < 4; ++tt)
    bptr[tt] = (const bf16x8*)&Bs[tcb + tt * 16 + (lane & 15)][bkh];

  for (int k0 = 0; k0 < K; k0 += BK) {
    if (fullTile) {
      // ---- A tile 64x32: float4 loads (global_load_b128), b64 LDS stores ----
      #pragma unroll
      for (int j = 0; j < 2; ++j) {
        int idx = tid + 256 * j;                 // 0..511 float4s
        int r = idx >> 3, c4 = (idx & 7) << 2;   // 8 float4 per row
        float4 av = *(const float4*)(A + (size_t)(rowBase + r) * K + k0 + c4);
        unsigned long long p =
            (unsigned long long)((unsigned)f2bf(av.x) | ((unsigned)f2bf(av.y) << 16)) |
            ((unsigned long long)((unsigned)f2bf(av.z) | ((unsigned)f2bf(av.w) << 16)) << 32);
        *(unsigned long long*)&As[r][c4] = p;
      }
      // ---- B tile 32x128 -> transposed LDS. k-pair b32 loads (always legal:
      //      lm_W rows are only 4B-aligned), packed b32 LDS stores. ----
      #pragma unroll
      for (int j = 0; j < 8; ++j) {
        int pidx = tid + 256 * j;                // 0..2047 = 16 kpairs x 128 n
        int kp = pidx >> 7;                      // k-pair 0..15
        int n  = pidx & 127;
        const float* bp = Bm + (size_t)(k0 + 2 * kp) * N + colBase + n;
        unsigned int lo = f2bf(bp[0]);
        unsigned int hi = f2bf(bp[N]);
        *(unsigned int*)&Bs[n][2 * kp] = lo | (hi << 16);
      }
    } else {
      // ---- guarded tail path (only last logits block-column) ----
      #pragma unroll
      for (int j = 0; j < 8; ++j) {              // A: 2048 elems
        int idx = tid + 256 * j;
        int r = idx >> 5, c = idx & 31;
        int gr = rowBase + r, gk = k0 + c;
        float vv = (gr < M && gk < K) ? A[(size_t)gr * K + gk] : 0.0f;
        As[r][c] = f2bf(vv);
      }
      #pragma unroll
      for (int j = 0; j < 16; ++j) {             // B: 4096 elems
        int idx = tid + 256 * j;
        int r = idx >> 7, c = idx & 127;         // r = k row, c = n col
        int gk = k0 + r, gc = colBase + c;
        float vv = (gk < K && gc < N) ? Bm[(size_t)gk * N + gc] : 0.0f;
        Bs[c][r] = f2bf(vv);
      }
    }
    __syncthreads();

    // ---- prefetch next K tile (global_prefetch_b8) ----
    if (k0 + BK < K && fullTile) {
      __builtin_prefetch(A + (size_t)(rowBase + (tid & 63)) * K + k0 + BK, 0, 1);
      __builtin_prefetch(Bm + (size_t)(k0 + BK + (tid & 31)) * N + colBase + ((tid >> 5) << 4), 0, 1);
    }

    // ---- fragments: two ds_load_b128 each; all loads hoisted before MACs ----
    bf16x8 alo = *aptr0;
    bf16x8 ahi = *aptr1;
    bf16x16 afrag = __builtin_shufflevector(alo, ahi,
        0, 1, 2, 3, 4, 5, 6, 7, 8, 9, 10, 11, 12, 13, 14, 15);

    bf16x16 bfr[4];
    #pragma unroll
    for (int tt = 0; tt < 4; ++tt) {
      bf16x8 blo = bptr[tt][0];
      bf16x8 bhi = bptr[tt][1];
      bfr[tt] = __builtin_shufflevector(blo, bhi,
          0, 1, 2, 3, 4, 5, 6, 7, 8, 9, 10, 11, 12, 13, 14, 15);
    }
    #pragma unroll
    for (int tt = 0; tt < 4; ++tt)
      acc[tt] = __builtin_amdgcn_wmma_f32_16x16x32_bf16(false, afrag, false, bfr[tt],
                                                        (short)0, acc[tt], false, false);
    __syncthreads();
  }

  // ---- epilogue: C/D layout = VGPR i -> M = i (+8 for lanes 16..31) ----
  const int mSub = tr * 16 + (lane >> 4) * 8;
  #pragma unroll
  for (int tt = 0; tt < 4; ++tt) {
    int n = colBase + tcb + tt * 16 + (lane & 15);
    if (n >= N) continue;
    float bval = bias ? bias[n] : 0.0f;
    #pragma unroll
    for (int i = 0; i < 8; ++i) {
      int gr = rowBase + mSub + i;
      if (gr >= M) continue;
      float vv = acc[tt][i] + bval;
      if (relu) vv = fmaxf(vv, 0.0f);
      if (res)  vv += res[(size_t)gr * N + n];
      C[(size_t)gr * N + n] = vv;
    }
  }
}

// ---------------- Embedding: x = tok_emb[idx] + pos_emb ----------------------
__global__ __launch_bounds__(256)
void embed_k(const int* __restrict__ idx, const float* __restrict__ tok,
             const float* __restrict__ pos, float* __restrict__ x)
{
  int gid = blockIdx.x * 256 + threadIdx.x;
  if (gid >= BT * D_MODEL) return;
  int bt = gid / D_MODEL;
  int d  = gid - bt * D_MODEL;
  int t  = bt & (T_SEQ - 1);
  x[gid] = tok[(size_t)idx[bt] * D_MODEL + d] + pos[(size_t)t * D_MODEL + d];
}

// ---------------- LayerNorm (one block per row) ------------------------------
__global__ __launch_bounds__(256)
void layernorm_k(const float* __restrict__ x, const float* __restrict__ s,
                 const float* __restrict__ b, float* __restrict__ y)
{
  __shared__ float red[256];
  __shared__ float red2[256];
  const int row = blockIdx.x;
  const float* xr = x + (size_t)row * D_MODEL;
  float lsum = 0.0f, lsq = 0.0f;
  for (int d = threadIdx.x; d < D_MODEL; d += 256) {
    float v = xr[d]; lsum += v; lsq += v * v;
  }
  red[threadIdx.x] = lsum; red2[threadIdx.x] = lsq;
  __syncthreads();
  for (int off = 128; off > 0; off >>= 1) {
    if (threadIdx.x < off) {
      red[threadIdx.x]  += red[threadIdx.x + off];
      red2[threadIdx.x] += red2[threadIdx.x + off];
    }
    __syncthreads();
  }
  const float mean = red[0]  * (1.0f / D_MODEL);
  const float var  = red2[0] * (1.0f / D_MODEL) - mean * mean;
  const float inv  = rsqrtf(var + EPS_LN);
  float* yr = y + (size_t)row * D_MODEL;
  for (int d = threadIdx.x; d < D_MODEL; d += 256)
    yr[d] = (xr[d] - mean) * inv * s[d] + b[d];
}

// ---------------- LDS-tiled streaming causal attention ----------------------
// One 128-thread block = 128 consecutive query rows of one (b,h).
// k/v staged through LDS in 32-row tiles: each k/v byte is read from L2 once
// per 128 queries. Tile staging uses GLOBAL_LOAD_ASYNC_TO_LDS_B128 (ASYNCcnt)
// when the toolchain exposes it; otherwise a synchronous float4 copy.
#define ATT_TQ 128
#define ATT_TK 32

__global__ __launch_bounds__(128)
void attention_k(const float* __restrict__ q, const float* __restrict__ k,
                 const float* __restrict__ v, float* __restrict__ o)
{
  __shared__ float ks[ATT_TK][HS_HEAD];   // 8 KB
  __shared__ float vs[ATT_TK][HS_HEAD];   // 8 KB

  const int tid   = threadIdx.x;
  const int tqi   = blockIdx.x % (T_SEQ / ATT_TQ);   // query chunk in sequence
  const int bh    = blockIdx.x / (T_SEQ / ATT_TQ);   // b*H + h
  const int h     = bh % H_HEADS;
  const int b     = bh / H_HEADS;
  const int t     = tqi * ATT_TQ + tid;              // this thread's query row

  const float* qr_p = q + ((size_t)(b * T_SEQ + t)) * D_MODEL + h * HS_HEAD;
  float4 qv[HS_HEAD / 4];
  #pragma unroll
  for (int i = 0; i < HS_HEAD / 4; ++i) qv[i] = ((const float4*)qr_p)[i];

  const float* kb = k + (size_t)b * T_SEQ * D_MODEL + h * HS_HEAD;
  const float* vb = v + (size_t)b * T_SEQ * D_MODEL + h * HS_HEAD;

  float m = -INFINITY, l = 0.0f;
  float acc[HS_HEAD];
  #pragma unroll
  for (int e = 0; e < HS_HEAD; ++e) acc[e] = 0.0f;
  const float scale = 0.125f;               // HS^-0.5 = 1/8

  const int nTiles = (tqi * ATT_TQ + ATT_TQ) / ATT_TK;  // covers all s <= t
  for (int kt = 0; kt < nTiles; ++kt) {
    const int kt0 = kt * ATT_TK;
    // cooperative tile load: 32 rows x 64 floats for k and v
#if USE_ASYNC_LDS
    #pragma unroll
    for (int j = 0; j < 4; ++j) {
      int idx = tid + 128 * j;                  // 0..511 float4s
      int r = idx >> 4, c = (idx & 15) << 2;    // 16 float4 per row
      __builtin_amdgcn_global_load_async_to_lds_b128(
          (v4i_vs*)(kb + (size_t)(kt0 + r) * D_MODEL + c),
          (lds_v4i_p)&ks[r][c], 0, 0);
      __builtin_amdgcn_global_load_async_to_lds_b128(
          (v4i_vs*)(vb + (size_t)(kt0 + r) * D_MODEL + c),
          (lds_v4i_p)&vs[r][c], 0, 0);
    }
    __builtin_amdgcn_s_wait_asynccnt(0);
    __syncthreads();
#else
    #pragma unroll
    for (int j = 0; j < 4; ++j) {
      int idx = tid + 128 * j;                  // 0..511 float4s
      int r = idx >> 4, c = (idx & 15) << 2;    // 16 float4 per row
      *(float4*)&ks[r][c] = *(const float4*)(kb + (size_t)(kt0 + r) * D_MODEL + c);
      *(float4*)&vs[r][c] = *(const float4*)(vb + (size_t)(kt0 + r) * D_MODEL + c);
    }
    __syncthreads();
#endif

    const int sEnd = (kt0 + ATT_TK < t + 1) ? (kt0 + ATT_TK) : (t + 1);
    for (int s = kt0; s < sEnd; ++s) {
      const float4* kr = (const float4*)&ks[s - kt0][0];   // broadcast reads
      float dot = 0.0f;
      #pragma unroll
      for (int i = 0; i < HS_HEAD / 4; ++i) {
        float4 kk = kr[i];
        dot += qv[i].x * kk.x + qv[i].y * kk.y + qv[i].z * kk.z + qv[i].w * kk.w;
      }
      dot *= scale;
      const float mn   = fmaxf(m, dot);
      const float corr = (m == -INFINITY) ? 0.0f : __expf(m - mn);
      const float p    = __expf(dot - mn);
      l = l * corr + p;
      const float4* vr = (const float4*)&vs[s - kt0][0];
      #pragma unroll
      for (int i = 0; i < HS_HEAD / 4; ++i) {
        float4 vv = vr[i];
        acc[4*i+0] = acc[4*i+0] * corr + p * vv.x;
        acc[4*i+1] = acc[4*i+1] * corr + p * vv.y;
        acc[4*i+2] = acc[4*i+2] * corr + p * vv.z;
        acc[4*i+3] = acc[4*i+3] * corr + p * vv.w;
      }
      m = mn;
    }
    __syncthreads();
  }

  const float inv = 1.0f / l;
  float* orow = o + ((size_t)(b * T_SEQ + t)) * D_MODEL + h * HS_HEAD;
  #pragma unroll
  for (int e = 0; e < HS_HEAD; ++e) orow[e] = acc[e] * inv;
}

// ---------------- Loss: mean NLL via log-softmax (one block per row) --------
__global__ void zero_k(float* p) { if (threadIdx.x == 0 && blockIdx.x == 0) *p = 0.0f; }

__global__ __launch_bounds__(256)
void loss_k(const float* __restrict__ logits, const int* __restrict__ targets,
            float* __restrict__ loss)
{
  __shared__ float red[256];
  const int row = blockIdx.x;
  const float* lr = logits + (size_t)row * V_SZ;

  float lmax = -INFINITY;
  for (int j = threadIdx.x; j < V_SZ; j += 256) lmax = fmaxf(lmax, lr[j]);
  red[threadIdx.x] = lmax;
  __syncthreads();
  for (int off = 128; off > 0; off >>= 1) {
    if (threadIdx.x < off) red[threadIdx.x] = fmaxf(red[threadIdx.x], red[threadIdx.x + off]);
    __syncthreads();
  }
  const float gmax = red[0];
  __syncthreads();

  float lsum = 0.0f;
  for (int j = threadIdx.x; j < V_SZ; j += 256) lsum += __expf(lr[j] - gmax);
  red[threadIdx.x] = lsum;
  __syncthreads();
  for (int off = 128; off > 0; off >>= 1) {
    if (threadIdx.x < off) red[threadIdx.x] += red[threadIdx.x + off];
    __syncthreads();
  }
  if (threadIdx.x == 0) {
    const float lse = gmax + __logf(red[0]);
    const float nll = lse - lr[targets[row]];
    atomicAdd(loss, nll * (1.0f / (float)BT));
  }
}

// ---------------- Host orchestration ----------------------------------------
extern "C" void kernel_launch(void* const* d_in, const int* in_sizes, int n_in,
                              void* d_out, int out_size, void* d_ws, size_t ws_size,
                              hipStream_t stream)
{
  (void)in_sizes; (void)n_in; (void)out_size; (void)ws_size;

  const int*   idx     = (const int*)  d_in[0];
  const int*   targets = (const int*)  d_in[1];
  const float* tok     = (const float*)d_in[2];
  const float* pos     = (const float*)d_in[3];
  const float* ln1_s   = (const float*)d_in[4];
  const float* ln1_b   = (const float*)d_in[5];
  const float* Wq      = (const float*)d_in[6];
  const float* Wk      = (const float*)d_in[7];
  const float* Wv      = (const float*)d_in[8];
  const float* Wo      = (const float*)d_in[9];
  const float* bo      = (const float*)d_in[10];
  const float* ln2_s   = (const float*)d_in[11];
  const float* ln2_b   = (const float*)d_in[12];
  const float* W1      = (const float*)d_in[13];
  const float* b1      = (const float*)d_in[14];
  const float* W2      = (const float*)d_in[15];
  const float* b2      = (const float*)d_in[16];
  const float* lnf_s   = (const float*)d_in[17];
  const float* lnf_b   = (const float*)d_in[18];
  const float* lm_W    = (const float*)d_in[19];
  const float* lm_b    = (const float*)d_in[20];

  float* logits = (float*)d_out;
  float* loss   = logits + (size_t)BT * V_SZ;

  const size_t NBTD = (size_t)BT * D_MODEL;
  float* ws  = (float*)d_ws;
  float* x   = ws;                // residual stream   [BT, D]
  float* h   = x  + NBTD;         // LN output         [BT, D]
  float* q   = h  + NBTD;         // queries           [BT, D]
  float* k   = q  + NBTD;         // keys              [BT, D]
  float* v   = k  + NBTD;         // values            [BT, D]
  float* ao  = v  + NBTD;         // attention output  [BT, D]
  float* mlp = ao + NBTD;         // MLP hidden        [BT, 4D]

  auto gemm = [&](const float* A, const float* Bm, const float* bias,
                  const float* res, float* C, int M, int N, int K, int relu) {
    dim3 grid((N + BN - 1) / BN, (M + BM - 1) / BM);
    gemm_bf16_wmma<<<grid, 256, 0, stream>>>(A, Bm, bias, res, C, M, N, K, relu);
  };

  // x = tok_emb[idx] + pos_emb
  embed_k<<<(BT * D_MODEL + 255) / 256, 256, 0, stream>>>(idx, tok, pos, x);

  for (int l = 0; l < L_LAYERS; ++l) {
    const float* wq  = Wq + (size_t)l * D_MODEL * D_MODEL;
    const float* wk  = Wk + (size_t)l * D_MODEL * D_MODEL;
    const float* wv  = Wv + (size_t)l * D_MODEL * D_MODEL;
    const float* wo  = Wo + (size_t)l * D_MODEL * D_MODEL;
    const float* w1  = W1 + (size_t)l * D_MODEL * 4 * D_MODEL;
    const float* w2  = W2 + (size_t)l * 4 * D_MODEL * D_MODEL;

    layernorm_k<<<BT, 256, 0, stream>>>(x, ln1_s + (size_t)l * D_MODEL,
                                        ln1_b + (size_t)l * D_MODEL, h);
    gemm(h, wq, nullptr, nullptr, q, BT, D_MODEL, D_MODEL, 0);
    gemm(h, wk, nullptr, nullptr, k, BT, D_MODEL, D_MODEL, 0);
    gemm(h, wv, nullptr, nullptr, v, BT, D_MODEL, D_MODEL, 0);

    attention_k<<<B_BATCH * H_HEADS * (T_SEQ / ATT_TQ), ATT_TQ, 0, stream>>>(q, k, v, ao);

    // x = x + ao @ Wo + bo
    gemm(ao, wo, bo + (size_t)l * D_MODEL, x, x, BT, D_MODEL, D_MODEL, 0);

    layernorm_k<<<BT, 256, 0, stream>>>(x, ln2_s + (size_t)l * D_MODEL,
                                        ln2_b + (size_t)l * D_MODEL, h);
    // mlp = relu(h @ W1 + b1)
    gemm(h, w1, b1 + (size_t)l * 4 * D_MODEL, nullptr, mlp, BT, 4 * D_MODEL, D_MODEL, 1);
    // x = x + mlp @ W2 + b2
    gemm(mlp, w2, b2 + (size_t)l * D_MODEL, x, x, BT, D_MODEL, 4 * D_MODEL, 0);
  }

  layernorm_k<<<BT, 256, 0, stream>>>(x, lnf_s, lnf_b, h);
  // logits = h @ lm_W + lm_b   (N = 50257; tail block takes guarded path)
  gemm(h, lm_W, lm_b, nullptr, logits, BT, V_SZ, D_MODEL, 0);

  zero_k<<<1, 1, 0, stream>>>(loss);
  loss_k<<<BT, 256, 0, stream>>>(logits, targets, loss);
}
